// PowerWhitening_592705487429
// MI455X (gfx1250) — compile-verified, matrix operand order checked by reading, not run on hardware
//
#include <hip/hip_runtime.h>
#include <hip/hip_bf16.h>

typedef __attribute__((ext_vector_type(16))) __bf16 v16bf;
typedef __attribute__((ext_vector_type(8)))  float  v8f;

#define NROWS 131072
#define DIM   256

__device__ __forceinline__ unsigned bf16r(float f) {
    // round-to-nearest-even f32 -> bf16 (returned in low 16 bits)
    unsigned u = __builtin_bit_cast(unsigned, f);
    u += 0x7fffu + ((u >> 16) & 1u);
    return u >> 16;
}

union ABits { unsigned u[8]; v16bf v; };

// ---------------------------------------------------------------- zero init
__global__ void pw_init_kernel(float* __restrict__ p, int n) {
    int i = blockIdx.x * 256 + threadIdx.x;
    if (i < n) p[i] = 0.0f;
}

// ---------------------------------------------------------------- column sums
__global__ void pw_colsum_kernel(const float* __restrict__ x, float* __restrict__ msum) {
    const int d = threadIdx.x;
    const int base = blockIdx.x * 512 * DIM;
    float s = 0.0f;
    for (int r = 0; r < 512; ++r) s += x[base + r * DIM + d];
    atomicAdd(&msum[d], s);
}

// ---------------------------------------------------------------- Gram = x^T x (bf16 WMMA, f32 accum)
// grid.x = 32 N-chunks (4096 rows each), grid.y = 4 (2x2 blocks of 128x128)
__global__ void pw_gram_kernel(const float* __restrict__ x, float* __restrict__ gram) {
    __shared__ __align__(16) unsigned short XA[128 * 32];
    __shared__ __align__(16) unsigned short XB[128 * 32];
    const int t    = threadIdx.x;
    const int wv   = t >> 5;
    const int lane = t & 31;
    const int half = lane >> 4;
    const int mn   = lane & 15;
    const int rb   = (int)(blockIdx.y >> 1) * 128;
    const int cb   = (int)(blockIdx.y & 1) * 128;
    const int n0   = (int)blockIdx.x * 4096;
    const int srow = t >> 3;          // 0..31 (k within chunk step)
    const int scol = (t & 7) * 16;    // 0..112

    v8f acc[8] = {};

    for (int ks = 0; ks < 128; ++ks) {
        const int nbase = n0 + ks * 32;
        // stage 32 rows x 128 cols (both operand column ranges) as bf16, [d][k] layout
        {
            const float* pa = x + (nbase + srow) * DIM + rb + scol;
            const float* pb = x + (nbase + srow) * DIM + cb + scol;
#pragma unroll
            for (int i = 0; i < 16; ++i) {
                XA[(scol + i) * 32 + srow] = (unsigned short)bf16r(pa[i]);
                XB[(scol + i) * 32 + srow] = (unsigned short)bf16r(pb[i]);
            }
        }
        __syncthreads();

        // A tile for this wave's 16-row strip: per-lane K runs [half*8, +8) and [16+half*8, +8)
        ABits a;
        {
            const unsigned* w = (const unsigned*)XA;
            const int dloc = wv * 16 + mn;
            const int b0 = (dloc * 32 + half * 8) >> 1;
#pragma unroll
            for (int i = 0; i < 4; ++i) a.u[i] = w[b0 + i];
#pragma unroll
            for (int i = 0; i < 4; ++i) a.u[4 + i] = w[b0 + 8 + i];
        }
        const unsigned* wb = (const unsigned*)XB;
#pragma unroll
        for (int ct = 0; ct < 8; ++ct) {
            ABits b;
            const int dloc = ct * 16 + mn;
            const int b0 = (dloc * 32 + half * 8) >> 1;
#pragma unroll
            for (int i = 0; i < 4; ++i) b.u[i] = wb[b0 + i];
#pragma unroll
            for (int i = 0; i < 4; ++i) b.u[4 + i] = wb[b0 + 8 + i];
            acc[ct] = __builtin_amdgcn_wmma_f32_16x16x32_bf16(
                false, a.v, false, b.v, (short)0, acc[ct], false, false);
        }
        __syncthreads();
    }

    // D layout: lanes 0-15 -> M=j, lanes 16-31 -> M=8+j; N = lane%16
#pragma unroll
    for (int ct = 0; ct < 8; ++ct) {
        const int c = cb + ct * 16 + mn;
#pragma unroll
        for (int j = 0; j < 8; ++j) {
            const int r = rb + wv * 16 + half * 8 + j;
            atomicAdd(&gram[r * DIM + c], acc[ct][j]);
        }
    }
}

// ---------------------------------------------------------------- cov = G/N - m m^T (in place)
__global__ void pw_cov_kernel(float* __restrict__ gram, const float* __restrict__ msum) {
    const int i = blockIdx.x * 256 + threadIdx.x;
    const int r = i >> 8, c = i & 255;
    const float inN = 1.0f / (float)NROWS;
    gram[i] = gram[i] * inN - (msum[r] * inN) * (msum[c] * inN);
}

// ---------------------------------------------------------------- serial deflation power iteration
// single workgroup; A resident in LDS with 257-float pitch (bank-conflict free matvec)
__global__ void pw_power_kernel(const float* __restrict__ cov, const float* __restrict__ R,
                                float* __restrict__ Vg, float* __restrict__ lg) {
    extern __shared__ float sm[];
    float* A   = sm;                 // 256*257
    float* v   = sm + 256 * 257;     // 256
    float* red = v + 256;            // 256
    const int t = threadIdx.x;

    for (int i = t; i < DIM * DIM; i += 256)
        A[(i >> 8) * 257 + (i & 255)] = cov[i];
    __syncthreads();

    for (int i = 0; i < DIM; ++i) {
        v[t] = R[t * DIM + i];       // column i of R (= row i of R.T)
        __syncthreads();
        float s = 0.0f;
        for (int it = 0; it < 50; ++it) {
            s = 0.0f;
            const float* row = A + t * 257;
            for (int j = 0; j < DIM; ++j) s += row[j] * v[j];
            red[t] = s * s; __syncthreads();
            for (int off = 128; off > 0; off >>= 1) {
                if (t < off) red[t] += red[t + off];
                __syncthreads();
            }
            v[t] = s * rsqrtf(red[0]);
            __syncthreads();
        }
        // l = ||A v||
        s = 0.0f;
        {
            const float* row = A + t * 257;
            for (int j = 0; j < DIM; ++j) s += row[j] * v[j];
        }
        red[t] = s * s; __syncthreads();
        for (int off = 128; off > 0; off >>= 1) {
            if (t < off) red[t] += red[t + off];
            __syncthreads();
        }
        const float l  = sqrtf(red[0]);
        const float vt = v[t];
        Vg[i * DIM + t] = vt;
        if (t == 0) lg[i] = l;
        // deflate: A -= l v v^T
        float* row = A + t * 257;
        for (int j = 0; j < DIM; ++j) row[j] -= l * vt * v[j];
        __syncthreads();
    }
}

// ---------------------------------------------------------------- W row r (bf16) + corr = (mean @ W)
__global__ void pw_buildw_kernel(const float* __restrict__ Vg, const float* __restrict__ lg,
                                 const float* __restrict__ msum,
                                 unsigned short* __restrict__ Wb, float* __restrict__ corr) {
    __shared__ float sc[256];
    const int r = blockIdx.x, c = threadIdx.x;
    sc[c] = Vg[c * DIM + r] * rsqrtf(lg[c]);
    __syncthreads();
    float acc = 0.0f;
    for (int i = 0; i < DIM; ++i) acc += sc[i] * Vg[i * DIM + c];
    Wb[r * DIM + c] = (unsigned short)bf16r(acc);
    atomicAdd(&corr[c], (msum[r] * (1.0f / (float)NROWS)) * acc);
}

// ---------------------------------------------------------------- out = x @ W - corr  (W symmetric => B rows contiguous)
__global__ void pw_out_kernel(const float* __restrict__ x, const unsigned short* __restrict__ Wb,
                              const float* __restrict__ corr, float* __restrict__ out) {
    const int t    = threadIdx.x;
    const int wv   = t >> 5;
    const int lane = t & 31;
    const int half = lane >> 4;
    const int mn   = lane & 15;
    const int m0   = ((int)blockIdx.x * 8 + wv) * 16;   // 16-row strip per wave
    const int rowA = m0 + mn;
    const unsigned* W32 = (const unsigned*)Wb;

    v8f acc[16] = {};

#pragma unroll 1
    for (int kk = 0; kk < 8; ++kk) {
        const int k0 = kk * 32;
        ABits a;
        {
            const float* p1 = x + rowA * DIM + k0 + half * 8;
            const float* p2 = p1 + 16;
            if (kk < 7) __builtin_prefetch(p1 + 32, 0, 1);
#pragma unroll
            for (int i = 0; i < 4; ++i) a.u[i]     = (bf16r(p1[2 * i + 1]) << 16) | bf16r(p1[2 * i]);
#pragma unroll
            for (int i = 0; i < 4; ++i) a.u[4 + i] = (bf16r(p2[2 * i + 1]) << 16) | bf16r(p2[2 * i]);
        }
#pragma unroll
        for (int ct = 0; ct < 16; ++ct) {
            ABits b;
            const int d  = ct * 16 + mn;            // W row d == W column d (symmetric)
            const int b0 = (d * DIM + k0 + half * 8) >> 1;
#pragma unroll
            for (int i = 0; i < 4; ++i) b.u[i]     = W32[b0 + i];
#pragma unroll
            for (int i = 0; i < 4; ++i) b.u[4 + i] = W32[b0 + 8 + i];
            acc[ct] = __builtin_amdgcn_wmma_f32_16x16x32_bf16(
                false, a.v, false, b.v, (short)0, acc[ct], false, false);
        }
    }

#pragma unroll
    for (int ct = 0; ct < 16; ++ct) {
        const int c = ct * 16 + mn;
        const float cr = corr[c];
#pragma unroll
        for (int j = 0; j < 8; ++j) {
            const int row = m0 + half * 8 + j;
            out[row * DIM + c] = acc[ct][j] - cr;
        }
    }
}

// ---------------------------------------------------------------- launch
extern "C" void kernel_launch(void* const* d_in, const int* in_sizes, int n_in,
                              void* d_out, int out_size, void* d_ws, size_t ws_size,
                              hipStream_t stream) {
    const float* x = (const float*)d_in[0];   // [131072, 256]
    const float* R = (const float*)d_in[1];   // [256, 256]
    float* out = (float*)d_out;

    float* ws   = (float*)d_ws;
    float* msum = ws;                         // 256    column sums
    float* gram = ws + 256;                   // 65536  Gram -> cov (in place)
    float* Vg   = ws + 256 + 65536;           // 65536  eigenvectors (row i = v_i)
    float* lg   = Vg + 65536;                 // 256    eigenvalues
    float* corr = lg + 256;                   // 256    mean @ W
    unsigned short* Wb = (unsigned short*)(corr + 256);  // 65536 bf16 W

    // zero accumulators (msum + gram contiguous: 65792 floats; corr separate)
    pw_init_kernel<<<dim3(257), dim3(256), 0, stream>>>(ws, 256 + 65536);
    pw_init_kernel<<<dim3(1),   dim3(256), 0, stream>>>(corr, 256);

    pw_colsum_kernel<<<dim3(256), dim3(256), 0, stream>>>(x, msum);
    pw_gram_kernel<<<dim3(32, 4), dim3(256), 0, stream>>>(x, gram);
    pw_cov_kernel<<<dim3(256), dim3(256), 0, stream>>>(gram, msum);

    const size_t smem = (size_t)(256 * 257 + 512) * sizeof(float);  // ~265 KB of the 320 KB WGP LDS
    pw_power_kernel<<<dim3(1), dim3(256), smem, stream>>>(gram, R, Vg, lg);

    pw_buildw_kernel<<<dim3(256), dim3(256), 0, stream>>>(Vg, lg, msum, Wb, corr);
    pw_out_kernel<<<dim3(1024), dim3(256), 0, stream>>>(x, Wb, corr, out);
}
